// HP_PPI_Model_25391846654580
// MI455X (gfx1250) — compile-verified, matrix operand order checked by reading, not run on hardware
//
#include <hip/hip_runtime.h>
#include <hip/hip_bf16.h>
#include <stdint.h>

// ---------------------------------------------------------------------------
// Types for CDNA5 WMMA (wave32): D(16x16 f32) = A(16x32 bf16) x B(32x16 bf16) + C
// ---------------------------------------------------------------------------
typedef __attribute__((ext_vector_type(16))) __bf16   v16bf;
typedef __attribute__((ext_vector_type(4)))  __bf16   bf16x4;
typedef __attribute__((ext_vector_type(8)))  float    v8f;
typedef __attribute__((ext_vector_type(4)))  float    f32x4;
typedef __attribute__((ext_vector_type(8)))  uint16_t u16x8;

#define F_IN    1024
#define H       64

// Native hardware conversion (v_cvt_*bf16_f32) instead of bit-twiddled RNE.
static __device__ __forceinline__ __bf16 f2bf(float f) { return (__bf16)f; }

// A fragment: 16x32 bf16, lane holds row M = lane%16, half = lane/16.
// element i (0..15): k = 16*(i/8) + 8*half + (i%8)   (ISA 7.12.2 16-bit A layout)
static __device__ __forceinline__ v16bf load_a_frag(const float* __restrict__ row,
                                                    int kg, int half, float scale) {
  const float* p0 = row + kg + 8 * half;       // k = kg + 8*half + 0..7
  const float* p1 = p0 + 16;                   // k = kg + 16 + 8*half + 0..7
  f32x4 q0 = *(const f32x4*)(p0);
  f32x4 q1 = *(const f32x4*)(p0 + 4);
  f32x4 q2 = *(const f32x4*)(p1);
  f32x4 q3 = *(const f32x4*)(p1 + 4);
  union { bf16x4 q[4]; v16bf v; } u;
  u.q[0] = __builtin_convertvector(q0 * scale, bf16x4);   // packed cvt pairs
  u.q[1] = __builtin_convertvector(q1 * scale, bf16x4);
  u.q[2] = __builtin_convertvector(q2 * scale, bf16x4);
  u.q[3] = __builtin_convertvector(q3 * scale, bf16x4);
  return u.v;
}

// B fragment: 32x16 bf16, lane holds col N = lane%16; base points at Wt[n] + kstep.
// Same k-interleave as A; two 16B LDS reads (ds_load_b128).
static __device__ __forceinline__ v16bf load_b_frag(const __bf16* __restrict__ base, int half) {
  union { u16x8 h[2]; v16bf v; } u;
  u.h[0] = *(const u16x8*)(base + 8 * half);
  u.h[1] = *(const u16x8*)(base + 16 + 8 * half);
  return u.v;
}

static __device__ __forceinline__ v8f wmma_bf16(v16bf a, v16bf b, v8f c) {
  return __builtin_amdgcn_wmma_f32_16x16x32_bf16(false, a, false, b, (short)0, c, false, false);
}

// ---------------------------------------------------------------------------
// Kernel 1: h = X[nrows x 1024] @ W[1024 x 64] + bias    (WMMA, bf16 inputs)
// Block: 256 threads = 8 waves; wave -> 16-row x 64-col tile; block -> 128 rows.
// W chunk (128 K-rows) staged transposed in LDS as bf16.
// ---------------------------------------------------------------------------
__global__ __launch_bounds__(256) void gemm_in_kernel(
    const float* __restrict__ X, const float* __restrict__ W,
    const float* __restrict__ bias, float* __restrict__ OUT, int nrows) {
  __shared__ __align__(16) __bf16 ldsW[H * 128];  // Wt chunk: [n][128 k] (16 KB)
  const int tid  = threadIdx.x;
  const int wave = tid >> 5;
  const int lane = tid & 31;
  const int half = lane >> 4;
  const int lrow = lane & 15;
  const int rowBase = blockIdx.x * 128 + wave * 16;
  int arow = rowBase + lrow;
  if (arow >= nrows) arow = nrows - 1;               // clamp; EXEC stays all-1s
  const float* xrow = X + (size_t)arow * F_IN;

  v8f zero = {0.f, 0.f, 0.f, 0.f, 0.f, 0.f, 0.f, 0.f};
  v8f acc[4] = {zero, zero, zero, zero};

  for (int chunk = 0; chunk < F_IN / 128; ++chunk) {
    const int kc = chunk * 128;
    __syncthreads();
    for (int i = tid; i < H * 128; i += 256) {       // stage W^T chunk -> LDS bf16
      int kl = i >> 6;                               // 0..127
      int n  = i & 63;
      ldsW[n * 128 + kl] = f2bf(W[(size_t)(kc + kl) * H + n]);
    }
    __syncthreads();
#pragma unroll
    for (int ks = 0; ks < 4; ++ks) {                 // 4 K-steps of 32
      v16bf a = load_a_frag(xrow, kc + ks * 32, half, 1.0f);
#pragma unroll
      for (int nt = 0; nt < 4; ++nt) {
        int n = nt * 16 + lrow;
        v16bf b = load_b_frag(ldsW + n * 128 + ks * 32, half);
        acc[nt] = wmma_bf16(a, b, acc[nt]);
      }
    }
  }
#pragma unroll
  for (int nt = 0; nt < 4; ++nt) {
    int n = nt * 16 + lrow;
    float bv = bias[n];
#pragma unroll
    for (int r = 0; r < 8; ++r) {                    // C layout: M = r + 8*half
      int m = rowBase + r + 8 * half;
      if (m < nrows) OUT[(size_t)m * H + n] = acc[nt][r] + bv;
    }
  }
}

// ---------------------------------------------------------------------------
// Kernel 2: per-node in/out degree counts (identical across both layers)
// ---------------------------------------------------------------------------
__global__ __launch_bounds__(256) void count_kernel(
    const int* __restrict__ e_src, const int* __restrict__ e_dst,
    float* __restrict__ cm, float* __restrict__ cv, int nedge) {
  int i = blockIdx.x * 256 + threadIdx.x;
  if (i < nedge) {
    atomicAdd(&cm[e_src[i]], 1.0f);
    atomicAdd(&cv[e_dst[i]], 1.0f);
  }
}

// ---------------------------------------------------------------------------
// Kernel 3: scatter-add  AGG[dst[e]][f] += H[src[e]][f]   (64 lanes per edge)
// ---------------------------------------------------------------------------
__global__ __launch_bounds__(256) void scatter_kernel(
    const float* __restrict__ Hs, const int* __restrict__ SRC,
    const int* __restrict__ DST, float* __restrict__ AGG, int nedge) {
  const int f     = threadIdx.x & 63;
  const int esub  = threadIdx.x >> 6;                // 4 edges per pass
  const int ebase = blockIdx.x * 32;
#pragma unroll
  for (int it = 0; it < 8; ++it) {
    int e = ebase + it * 4 + esub;
    if (e < nedge) {
      int s = SRC[e], d = DST[e];
      atomicAdd(&AGG[(size_t)d * H + f], Hs[(size_t)s * H + f]);
    }
  }
}

// ---------------------------------------------------------------------------
// Kernel 4: SAGE linear  out = relu( (agg/deg) @ Wl + xdst @ Wr + b )
// mean-divide folded into A-fragment conversion (scale per row).
// ---------------------------------------------------------------------------
__global__ __launch_bounds__(256) void sage_kernel(
    const float* __restrict__ AGG, const float* __restrict__ CNT,
    const float* __restrict__ XD,
    const float* __restrict__ Wl, const float* __restrict__ Wr,
    const float* __restrict__ bias, float* __restrict__ OUT, int nrows) {
  __shared__ __align__(16) __bf16 ldsWl[H * H];      // 8 KB each, transposed [n][k]
  __shared__ __align__(16) __bf16 ldsWr[H * H];
  const int tid = threadIdx.x;
  for (int i = tid; i < H * H; i += 256) {
    int k = i >> 6, n = i & 63;
    ldsWl[n * H + k] = f2bf(Wl[i]);
    ldsWr[n * H + k] = f2bf(Wr[i]);
  }
  __syncthreads();

  const int wave = tid >> 5;
  const int lane = tid & 31;
  const int half = lane >> 4;
  const int lrow = lane & 15;
  const int rowBase = blockIdx.x * 128 + wave * 16;
  int arow = rowBase + lrow;
  if (arow >= nrows) arow = nrows - 1;
  const float deg    = CNT[arow];
  const float scale  = 1.0f / fmaxf(deg, 1.0f);
  const float* aggrow = AGG + (size_t)arow * H;
  const float* xrow   = XD  + (size_t)arow * H;

  v8f zero = {0.f, 0.f, 0.f, 0.f, 0.f, 0.f, 0.f, 0.f};
  v8f acc[4] = {zero, zero, zero, zero};

#pragma unroll
  for (int ks = 0; ks < 2; ++ks) {                   // (agg/deg) @ Wl
    v16bf a = load_a_frag(aggrow, ks * 32, half, scale);
#pragma unroll
    for (int nt = 0; nt < 4; ++nt) {
      int n = nt * 16 + lrow;
      acc[nt] = wmma_bf16(a, load_b_frag(ldsWl + n * H + ks * 32, half), acc[nt]);
    }
  }
#pragma unroll
  for (int ks = 0; ks < 2; ++ks) {                   // xdst @ Wr
    v16bf a = load_a_frag(xrow, ks * 32, half, 1.0f);
#pragma unroll
    for (int nt = 0; nt < 4; ++nt) {
      int n = nt * 16 + lrow;
      acc[nt] = wmma_bf16(a, load_b_frag(ldsWr + n * H + ks * 32, half), acc[nt]);
    }
  }
#pragma unroll
  for (int nt = 0; nt < 4; ++nt) {
    int n = nt * 16 + lrow;
    float bv = bias[n];
#pragma unroll
    for (int r = 0; r < 8; ++r) {
      int m = rowBase + r + 8 * half;
      if (m < nrows) OUT[(size_t)m * H + n] = fmaxf(acc[nt][r] + bv, 0.0f);
    }
  }
}

// ---------------------------------------------------------------------------
// Kernel 5: classifier  out[i] = concat(m2[el_src[i]], v2[el_dst[i]]) @ Wc + bc
// ---------------------------------------------------------------------------
__global__ __launch_bounds__(256) void cls_kernel(
    const float* __restrict__ m2, const float* __restrict__ v2,
    const int* __restrict__ el_src, const int* __restrict__ el_dst,
    const float* __restrict__ Wc, const float* __restrict__ bc,
    float* __restrict__ out, int n) {
  __shared__ float w[2 * H * 2];                     // 128x2
  __shared__ float bs[2];
  int tid = threadIdx.x;
  if (tid < 256) w[tid] = Wc[tid];
  if (tid < 2) bs[tid] = bc[tid];
  __syncthreads();
  int i = blockIdx.x * 256 + tid;
  if (i >= n) return;
  const float* a = m2 + (size_t)el_src[i] * H;
  const float* v = v2 + (size_t)el_dst[i] * H;
  float a0 = bs[0], a1 = bs[1];
#pragma unroll 4
  for (int k = 0; k < H; ++k) {
    float x = a[k];
    a0 += x * w[2 * k];
    a1 += x * w[2 * k + 1];
  }
#pragma unroll 4
  for (int k = 0; k < H; ++k) {
    float x = v[k];
    a0 += x * w[2 * (H + k)];
    a1 += x * w[2 * (H + k) + 1];
  }
  out[(size_t)i * 2 + 0] = a0;
  out[(size_t)i * 2 + 1] = a1;
}

// ---------------------------------------------------------------------------
// Host orchestration
// ---------------------------------------------------------------------------
extern "C" void kernel_launch(void* const* d_in, const int* in_sizes, int n_in,
                              void* d_out, int out_size, void* d_ws, size_t ws_size,
                              hipStream_t stream) {
  const float* x_mouse = (const float*)d_in[0];
  const float* x_virus = (const float*)d_in[1];
  const int*   e_src   = (const int*)d_in[2];
  const int*   e_dst   = (const int*)d_in[3];
  const int*   el_src  = (const int*)d_in[4];
  const int*   el_dst  = (const int*)d_in[5];
  const float* W_mouse = (const float*)d_in[6];
  const float* b_mouse = (const float*)d_in[7];
  const float* W_virus = (const float*)d_in[8];
  const float* b_virus = (const float*)d_in[9];
  const float* l1_mv_Wl = (const float*)d_in[10];
  const float* l1_mv_Wr = (const float*)d_in[11];
  const float* l1_mv_b  = (const float*)d_in[12];
  const float* l1_vm_Wl = (const float*)d_in[13];
  const float* l1_vm_Wr = (const float*)d_in[14];
  const float* l1_vm_b  = (const float*)d_in[15];
  const float* l2_mv_Wl = (const float*)d_in[16];
  const float* l2_mv_Wr = (const float*)d_in[17];
  const float* l2_mv_b  = (const float*)d_in[18];
  const float* l2_vm_Wl = (const float*)d_in[19];
  const float* l2_vm_Wr = (const float*)d_in[20];
  const float* l2_vm_b  = (const float*)d_in[21];
  const float* W_cls    = (const float*)d_in[22];
  const float* b_cls    = (const float*)d_in[23];

  const int NM = in_sizes[0] / F_IN;   // 50000
  const int NV = in_sizes[1] / F_IN;   // 20000
  const int NE = in_sizes[2];          // 2,000,000
  const int NL = in_sizes[4];          // 500,000

  // workspace carve-up (floats)
  float* ws    = (float*)d_ws;
  float* h_m   = ws;                       ws += (size_t)NM * H;
  float* h_v   = ws;                       ws += (size_t)NV * H;
  float* m1    = ws;                       ws += (size_t)NM * H;
  float* v1    = ws;                       ws += (size_t)NV * H;
  float* m2    = ws;                       ws += (size_t)NM * H;
  float* v2    = ws;                       ws += (size_t)NV * H;
  float* agg_m = ws;                       ws += (size_t)NM * H;
  float* agg_v = ws;                       ws += (size_t)NV * H;
  float* cnt_m = ws;                       ws += NM;
  float* cnt_v = ws;                       ws += NV;

  const int gM = (NM + 127) / 128;   // GEMM/SAGE blocks (128 rows each)
  const int gV = (NV + 127) / 128;
  const int gE32  = (NE + 31) / 32;
  const int gE256 = (NE + 255) / 256;
  const int gL    = (NL + 255) / 256;

  // 1. input projections (WMMA)
  gemm_in_kernel<<<gM, 256, 0, stream>>>(x_mouse, W_mouse, b_mouse, h_m, NM);
  gemm_in_kernel<<<gV, 256, 0, stream>>>(x_virus, W_virus, b_virus, h_v, NV);

  // 2. degree counts (shared by both layers)
  hipMemsetAsync(cnt_m, 0, (size_t)NM * sizeof(float), stream);
  hipMemsetAsync(cnt_v, 0, (size_t)NV * sizeof(float), stream);
  count_kernel<<<gE256, 256, 0, stream>>>(e_src, e_dst, cnt_m, cnt_v, NE);

  // 3. layer 1: mouse->virus
  hipMemsetAsync(agg_v, 0, (size_t)NV * H * sizeof(float), stream);
  scatter_kernel<<<gE32, 256, 0, stream>>>(h_m, e_src, e_dst, agg_v, NE);
  sage_kernel<<<gV, 256, 0, stream>>>(agg_v, cnt_v, h_v, l1_mv_Wl, l1_mv_Wr, l1_mv_b, v1, NV);
  //    layer 1: virus->mouse (reverse edges)
  hipMemsetAsync(agg_m, 0, (size_t)NM * H * sizeof(float), stream);
  scatter_kernel<<<gE32, 256, 0, stream>>>(h_v, e_dst, e_src, agg_m, NE);
  sage_kernel<<<gM, 256, 0, stream>>>(agg_m, cnt_m, h_m, l1_vm_Wl, l1_vm_Wr, l1_vm_b, m1, NM);

  // 4. layer 2
  hipMemsetAsync(agg_v, 0, (size_t)NV * H * sizeof(float), stream);
  scatter_kernel<<<gE32, 256, 0, stream>>>(m1, e_src, e_dst, agg_v, NE);
  sage_kernel<<<gV, 256, 0, stream>>>(agg_v, cnt_v, v1, l2_mv_Wl, l2_mv_Wr, l2_mv_b, v2, NV);

  hipMemsetAsync(agg_m, 0, (size_t)NM * H * sizeof(float), stream);
  scatter_kernel<<<gE32, 256, 0, stream>>>(v1, e_dst, e_src, agg_m, NE);
  sage_kernel<<<gM, 256, 0, stream>>>(agg_m, cnt_m, m1, l2_vm_Wl, l2_vm_Wr, l2_vm_b, m2, NM);

  // 5. classifier over label edges
  cls_kernel<<<gL, 256, 0, stream>>>(m2, v2, el_src, el_dst, W_cls, b_cls,
                                     (float*)d_out, NL);
}
